// MaskedSelfAttention_52999896432836
// MI455X (gfx1250) — matrix operand order repaired and checked by reading, not compiled, not run on hardware
//
#include <hip/hip_runtime.h>
#include <stdint.h>

// ---------------------------------------------------------------------------
// Masked self-attention for MI455X (gfx1250), wave32 + WMMA bf16.
// B=8, S=2048, E=1024, H=64. fp32 in/out, bf16 WMMA operands, fp32 accum.
//
// Roofline: ~15 GFLOP total vs 64 MB mandatory x-traffic at 23.3 TB/s
// => HBM-bound. Strategy: bf16 WMMA (fp32 accum), flash-attention online
// softmax (never materialize the 128 MB score matrix), and all operands
// pre-packed so every WMMA fragment is a contiguous 16/32 B load per lane.
// ---------------------------------------------------------------------------

typedef __attribute__((ext_vector_type(16))) __bf16        v16bf;
typedef __attribute__((ext_vector_type(8)))  float         v8f;
typedef __attribute__((ext_vector_type(8)))  unsigned int  v8u;

#define EMB   1024
#define HD    64
#define BATCH 8
#define SEQ   2048

#define WMMA_BF16(a, b, c) \
  __builtin_amdgcn_wmma_f32_16x16x32_bf16(false, (a), false, (b), (short)0, (c), false, false)

// fp32 -> bf16 via native conversion (v_cvt_pk_bf16_f32 on CDNA5)
__device__ __forceinline__ unsigned short f2bf_u(float f) {
  return __builtin_bit_cast(unsigned short, (__bf16)f);
}
__device__ __forceinline__ unsigned int pk_bf16(float a, float b) {
  return (unsigned)f2bf_u(a) | ((unsigned)f2bf_u(b) << 16);
}

__device__ __forceinline__ v8f splat8(float v) {
  v8f r;
#pragma unroll
  for (int i = 0; i < 8; ++i) r[i] = v;
  return r;
}

// A-fragment (16x32 bf16): per lane one contiguous 16-half run at p.
__device__ __forceinline__ v16bf ld_afrag(const unsigned short* p) {
  uint4 r0 = *(const uint4*)(p);
  uint4 r1 = *(const uint4*)(p + 8);
  v8u a;
  a[0] = r0.x; a[1] = r0.y; a[2] = r0.z; a[3] = r0.w;
  a[4] = r1.x; a[5] = r1.y; a[6] = r1.z; a[7] = r1.w;
  return __builtin_bit_cast(v16bf, a);
}

// B-fragment (32x16 bf16): per lane one contiguous 16-half (32 B) run.
__device__ __forceinline__ v16bf ld_bfrag(const unsigned short* p) {
  return __builtin_bit_cast(v16bf, *(const v8u*)p);
}

// Build A-fragment from 16 fp32 values (one contiguous 16-float run), native cvt.
__device__ __forceinline__ v16bf mk_a(float4 f0, float4 f1, float4 f2, float4 f3) {
  v16bf a;
  a[0]  = (__bf16)f0.x; a[1]  = (__bf16)f0.y; a[2]  = (__bf16)f0.z; a[3]  = (__bf16)f0.w;
  a[4]  = (__bf16)f1.x; a[5]  = (__bf16)f1.y; a[6]  = (__bf16)f1.z; a[7]  = (__bf16)f1.w;
  a[8]  = (__bf16)f2.x; a[9]  = (__bf16)f2.y; a[10] = (__bf16)f2.z; a[11] = (__bf16)f2.w;
  a[12] = (__bf16)f3.x; a[13] = (__bf16)f3.y; a[14] = (__bf16)f3.z; a[15] = (__bf16)f3.w;
  return a;
}

// 16-lane-half max reduction (rows of a C/D fragment live across 16 lanes).
__device__ __forceinline__ v8f redmax16(v8f v) {
#pragma unroll
  for (int d = 1; d < 16; d <<= 1) {
#pragma unroll
    for (int i = 0; i < 8; ++i) v[i] = fmaxf(v[i], __shfl_xor(v[i], d, 32));
  }
  return v;
}

// ---------------------------------------------------------------------------
// Kernel 1: repack Wq/Wk/Wv [E,H] fp32 -> bf16 in WMMA B-fragment layout.
// Layout: Wb[m][ ((kb*64 + col)*2 + half) * 16 + j ],  K = kb*32 + half*16 + j
// ---------------------------------------------------------------------------
__global__ void pack_w_kernel(const float* __restrict__ Wq,
                              const float* __restrict__ Wk,
                              const float* __restrict__ Wv,
                              unsigned short* __restrict__ Wb) {
  int t = blockIdx.x * blockDim.x + threadIdx.x;   // 3*32*64*2 = 12288 work items
  if (t >= 3 * 32 * 64 * 2) return;
  int m  = t >> 12;
  int r  = t & 4095;
  int kb = r >> 7;
  int r2 = r & 127;
  int c  = r2 >> 1;
  int h  = r2 & 1;
  const float* W = (m == 0) ? Wq : (m == 1) ? Wk : Wv;
  unsigned short* out = Wb + m * 65536 + (((kb * 64 + c) * 2 + h) << 4);
  int kBase = kb * 32 + h * 16;
#pragma unroll
  for (int j = 0; j < 16; ++j) out[j] = f2bf_u(W[(kBase + j) * HD + c]);
}

// ---------------------------------------------------------------------------
// Kernel 2: QKV projection. 1 block = 16 rows of x, 4 waves = 4 col slices.
// Emits: Qb/Kb row-major bf16 [B*S, 64]; Vp in B-fragment layout per batch:
//   Vp[b][ ((s/32)*64 + col)*32 + ((s%32)/16)*16 + (s%16) ]
// ---------------------------------------------------------------------------
__global__ void __launch_bounds__(128)
qkv_kernel(const float* __restrict__ x,
           const unsigned short* __restrict__ Wb,
           const float* __restrict__ bq, const float* __restrict__ bk,
           const float* __restrict__ bv,
           unsigned short* __restrict__ Qb, unsigned short* __restrict__ Kb,
           unsigned short* __restrict__ Vp) {
  const int wave = threadIdx.x >> 5;
  const int lane = threadIdx.x & 31;
  const int lo   = lane & 15;
  const bool hi  = lane >= 16;
  const int row0 = blockIdx.x * 16;            // global row in [0, B*S)

  v8f accQ = splat8(0.f), accK = splat8(0.f), accV = splat8(0.f);

  const int    kbase = hi ? 16 : 0;
  const float* xr    = x + (size_t)(row0 + lo) * EMB;
  const int    col   = wave * 16 + lo;
  const int    hB    = hi ? 1 : 0;

  for (int kb = 0; kb < 32; ++kb) {
    const int k0 = kb * 32;
    // speculative prefetch of next x tile into all cache levels (WGP scope)
    __builtin_prefetch(xr + k0 + 32, 0, 3);

    float4 f0 = *(const float4*)(xr + k0 + kbase);
    float4 f1 = *(const float4*)(xr + k0 + kbase + 4);
    float4 f2 = *(const float4*)(xr + k0 + kbase + 8);
    float4 f3 = *(const float4*)(xr + k0 + kbase + 12);
    v16bf a = mk_a(f0, f1, f2, f3);

    const unsigned short* wbase = Wb + (((kb * 64 + col) * 2 + hB) << 4);
    accQ = WMMA_BF16(a, ld_bfrag(wbase),             accQ);
    accK = WMMA_BF16(a, ld_bfrag(wbase + 65536),     accK);
    accV = WMMA_BF16(a, ld_bfrag(wbase + 2 * 65536), accV);
  }

  const float bQ = bq[col], bK = bk[col], bV = bv[col];
#pragma unroll
  for (int i = 0; i < 8; ++i) { accQ[i] += bQ; accK[i] += bK; accV[i] += bV; }

  // C/D layout: lanes 0-15 hold rows r, lanes 16-31 rows r+8.
  const int mofs = hi ? 8 : 0;
#pragma unroll
  for (int r = 0; r < 8; ++r) {
    const int s = row0 + r + mofs;
    Qb[(size_t)s * HD + col] = f2bf_u(accQ[r]);
    Kb[(size_t)s * HD + col] = f2bf_u(accK[r]);
  }
  // V -> packed B-fragment layout (one 16 B store per lane).
  const int b   = row0 >> 11;
  const int sb  = row0 & 2047;
  const int kbv = sb >> 5;
  const int hp  = (sb >> 4) & 1;
  unsigned short* vout =
      Vp + (size_t)b * SEQ * HD + ((kbv * 64 + col) * 32 + hp * 16 + mofs);
  *(uint4*)vout = make_uint4(pk_bf16(accV[0], accV[1]), pk_bf16(accV[2], accV[3]),
                             pk_bf16(accV[4], accV[5]), pk_bf16(accV[6], accV[7]));
}

// ---------------------------------------------------------------------------
// Kernel 3: flash attention. 1 wave = 16 query rows; key loop in 32-key blocks.
// ---------------------------------------------------------------------------
__global__ void __launch_bounds__(128)
attn_kernel(const unsigned short* __restrict__ Qb,
            const unsigned short* __restrict__ Kb,
            const unsigned short* __restrict__ Vp,
            const int* __restrict__ mask,
            float* __restrict__ out) {
  __shared__ unsigned short Plds[4][16][32];   // per-wave P tile (bf16)

  const int wave = threadIdx.x >> 5;
  const int lane = threadIdx.x & 31;
  const int lo   = lane & 15;
  const bool hi  = lane >= 16;
  const int b    = blockIdx.x >> 5;            // 32 query tiles of 64 per batch
  const int tile = blockIdx.x & 31;
  const int q0   = tile * 64 + wave * 16;      // query base within batch

  const unsigned short* Qb_b = Qb + (size_t)b * SEQ * HD;
  const unsigned short* Kb_b = Kb + (size_t)b * SEQ * HD;
  const unsigned short* Vp_b = Vp + (size_t)b * SEQ * HD;
  const int*            mk   = mask + b * SEQ;

  const int kbase = hi ? 16 : 0;
  const int hofs  = hi ? 16 : 0;

  // Q A-fragments for K-steps k0=0 and k0=32 (H=64).
  const unsigned short* qp = Qb_b + (size_t)(q0 + lo) * HD;
  v16bf aQ0 = ld_afrag(qp + 0  + kbase);
  v16bf aQ1 = ld_afrag(qp + 32 + kbase);

  // all-ones B-fragment: row-sum of P via one extra WMMA (replaces 32 shuffles)
  v16bf ones;
#pragma unroll
  for (int i = 0; i < 16; ++i) ones[i] = (__bf16)1.0f;

  v8f m = splat8(-INFINITY), l = splat8(0.f);
  v8f O0 = splat8(0.f), O1 = splat8(0.f), O2 = splat8(0.f), O3 = splat8(0.f);
  const float scale = 0.125f;                  // 1/sqrt(64)

  for (int kb2 = 0; kb2 < SEQ / 32; ++kb2) {
    const int key0 = kb2 * 32;
    // speculative prefetch of next K tile into all cache levels (WGP scope)
    __builtin_prefetch(Kb_b + (size_t)(key0 + 32 + lo) * HD, 0, 3);

    const int mv0 = mk[key0 + lo];
    const int mv1 = mk[key0 + 16 + lo];

    // S = Q * K^T  (two 16-key column tiles, two K-steps each)
    const unsigned short* kp0 = Kb_b + (size_t)(key0 + lo) * HD + hofs;
    const unsigned short* kp1 = Kb_b + (size_t)(key0 + 16 + lo) * HD + hofs;
    v8f s0 = splat8(0.f), s1 = splat8(0.f);
    s0 = WMMA_BF16(aQ0, ld_bfrag(kp0),      s0);
    s0 = WMMA_BF16(aQ1, ld_bfrag(kp0 + 32), s0);
    s1 = WMMA_BF16(aQ0, ld_bfrag(kp1),      s1);
    s1 = WMMA_BF16(aQ1, ld_bfrag(kp1 + 32), s1);

#pragma unroll
    for (int i = 0; i < 8; ++i) {
      s0[i] = (mv0 != 0) ? s0[i] * scale : -INFINITY;
      s1[i] = (mv1 != 0) ? s1[i] * scale : -INFINITY;
    }

    // online softmax: row max across the 32-key block
    v8f cm;
#pragma unroll
    for (int i = 0; i < 8; ++i) cm[i] = fmaxf(s0[i], s1[i]);
    cm = redmax16(cm);
    v8f mn, alpha, p0, p1;
#pragma unroll
    for (int i = 0; i < 8; ++i) {
      mn[i]    = fmaxf(m[i], cm[i]);
      alpha[i] = __expf(m[i] - mn[i]);
      p0[i]    = __expf(s0[i] - mn[i]);
      p1[i]    = __expf(s1[i] - mn[i]);
      m[i]     = mn[i];
    }

    // P (C/D layout) -> LDS -> A-fragment layout (wave-private tile; LDS ops
    // from one wave execute in order; compiler inserts s_wait_dscnt).
    const int prow = hi ? 8 : 0;
#pragma unroll
    for (int r = 0; r < 8; ++r) {
      Plds[wave][prow + r][lo]      = f2bf_u(p0[r]);
      Plds[wave][prow + r][16 + lo] = f2bf_u(p1[r]);
    }
    asm volatile("" ::: "memory");
    v16bf aP = ld_afrag(&Plds[wave][lo][kbase]);
    asm volatile("" ::: "memory");

    // row sums of (bf16-rounded) P via WMMA against all-ones B
    v8f rs = WMMA_BF16(aP, ones, splat8(0.f));
#pragma unroll
    for (int i = 0; i < 8; ++i) {
      l[i] = l[i] * alpha[i] + rs[i];
      O0[i] *= alpha[i]; O1[i] *= alpha[i]; O2[i] *= alpha[i]; O3[i] *= alpha[i];
    }

    // O += P * V   (4 output column tiles of 16)
    const unsigned short* vb = Vp_b + ((size_t)kb2 * 64 + lo) * 32 + hofs;
    O0 = WMMA_BF16(aP, ld_bfrag(vb),            O0);
    O1 = WMMA_BF16(aP, ld_bfrag(vb + 16 * 32),  O1);
    O2 = WMMA_BF16(aP, ld_bfrag(vb + 32 * 32),  O2);
    O3 = WMMA_BF16(aP, ld_bfrag(vb + 48 * 32),  O3);
  }

  // epilogue: normalize and store fp32
  v8f rl;
#pragma unroll
  for (int i = 0; i < 8; ++i) rl[i] = 1.0f / l[i];
  const int mofs = hi ? 8 : 0;
#pragma unroll
  for (int r = 0; r < 8; ++r) {
    const size_t row = (size_t)b * SEQ + q0 + mofs + r;
    out[row * HD + 0  + lo] = O0[r] * rl[r];
    out[row * HD + 16 + lo] = O1[r] * rl[r];
    out[row * HD + 32 + lo] = O2[r] * rl[r];
    out[row * HD + 48 + lo] = O3[r] * rl[r];
  }
}

// ---------------------------------------------------------------------------
extern "C" void kernel_launch(void* const* d_in, const int* in_sizes, int n_in,
                              void* d_out, int out_size, void* d_ws, size_t ws_size,
                              hipStream_t stream) {
  const float* x    = (const float*)d_in[0];
  const int*   mask = (const int*)d_in[1];
  const float* Wq   = (const float*)d_in[2];
  const float* bq   = (const float*)d_in[3];
  const float* Wk   = (const float*)d_in[4];
  const float* bk   = (const float*)d_in[5];
  const float* Wv   = (const float*)d_in[6];
  const float* bv   = (const float*)d_in[7];
  float*       out  = (float*)d_out;

  // workspace partition (needs ~6.5 MB)
  unsigned short* Qb = (unsigned short*)d_ws;                               // 2 MB
  unsigned short* Kb = (unsigned short*)((char*)d_ws + (2ull << 20));       // 2 MB
  unsigned short* Vp = (unsigned short*)((char*)d_ws + (4ull << 20));       // 2 MB
  unsigned short* Wb = (unsigned short*)((char*)d_ws + (6ull << 20));       // 384 KB

  hipLaunchKernelGGL(pack_w_kernel, dim3(48), dim3(256), 0, stream, Wq, Wk, Wv, Wb);
  hipLaunchKernelGGL(qkv_kernel, dim3((BATCH * SEQ) / 16), dim3(128), 0, stream,
                     x, Wb, bq, bk, bv, Qb, Kb, Vp);
  hipLaunchKernelGGL(attn_kernel, dim3(BATCH * (SEQ / 64)), dim3(128), 0, stream,
                     Qb, Kb, Vp, mask, out);
}
